// MultiHeadAttention_82523501625370
// MI455X (gfx1250) — matrix operand order repaired and checked by reading, not compiled, not run on hardware
//
#include <hip/hip_runtime.h>

// MHA w/ RoPE + sliding-window causal attention for gfx1250 (MI455X).
// Compute-bound (~172 GFLOP vs ~6us HBM traffic at 23.3 TB/s) -> bf16 WMMA
// (v_wmma_f32_16x16x32_bf16, fp32 accumulate). Weights and K are staged
// pre-transposed; WMMA B-fragments are gathered with strided scalar loads.
// Flash-attention K-tiles are fetched with the Tensor Data Mover
// (tensor_load_to_lds), double-buffered on TENSORcnt.

#define SEQ    2048
#define DM     2048
#define HEADS  16
#define HDIM   128
#define WINDOW 1024

typedef __attribute__((ext_vector_type(16))) __bf16 v16bf;
typedef __attribute__((ext_vector_type(8)))  float  v8f;
typedef __attribute__((ext_vector_type(4)))  unsigned int v4u;
typedef __attribute__((ext_vector_type(8)))  int          v8i;
typedef __attribute__((ext_vector_type(4)))  int          v4i;

union BF16x16 { v16bf v; uint4 u4[2]; __bf16 h[16]; };
union F32x8   { v8f v; float f[8]; };

__device__ __forceinline__ void compiler_fence() {
  asm volatile("" ::: "memory");
}

__device__ __forceinline__ void lds_sync() {
  __builtin_amdgcn_wave_barrier();
  asm volatile("s_wait_dscnt 0x0" ::: "memory");
  __builtin_amdgcn_wave_barrier();
}

__device__ __forceinline__ v8f wmma_bf16(v16bf a, v16bf b, v8f c) {
  // (neg_a, A, neg_b, B, c_mod, C, reuse_a, reuse_b)
  return __builtin_amdgcn_wmma_f32_16x16x32_bf16(false, a, false, b, (short)0, c,
                                                 false, false);
}

// Flat LDS pointers carry the LDS byte offset in addr[31:0] (ISA 10.2).
__device__ __forceinline__ unsigned lds_off_of(const void* p) {
  return (unsigned)(size_t)p;
}

// ---- Tensor Data Mover: 2D tile load (global, row-major, bf16) -> LDS ----
// D# encoding per CDNA5 ISA:
//   g0: count=1 | lds_addr | global_addr(57b) | type=2
//   g1: data_size=1(2B) | tensor_dim0/1 | tile_dim0/1 | tensor_dim0_stride
// This toolchain exposes the 6-arg builtin: (v4u, v8i, v4i, v4i, v8i, cpol).
__device__ __forceinline__ void tdm_load_2d(unsigned lds_byte_off,
                                            const void* gptr,
                                            unsigned tile_d0, unsigned tile_d1,
                                            unsigned long long stride_elems) {
  const unsigned long long ga = (unsigned long long)(size_t)gptr;
  v4u g0;
  g0[0] = 1u;                                        // count=1, user descriptor
  g0[1] = lds_byte_off;                              // lds_addr (bytes)
  g0[2] = (unsigned)(ga & 0xFFFFFFFFu);              // global_addr[31:0]
  g0[3] = (unsigned)((ga >> 32) & 0x01FFFFFFu) | (2u << 30);  // [56:32] | type=2
  v8i g1;
  g1[0] = (int)(1u << 16);                           // data_size=1 -> 2 bytes
  g1[1] = (int)((tile_d0 & 0xFFFFu) << 16);          // tensor_dim0[15:0]
  g1[2] = (int)(((tile_d0 >> 16) & 0xFFFFu) | ((tile_d1 & 0xFFFFu) << 16));
  g1[3] = (int)(((tile_d1 >> 16) & 0xFFFFu) | ((tile_d0 & 0xFFFFu) << 16)); // tile_dim0
  g1[4] = (int)(tile_d1 & 0xFFFFu);                  // tile_dim1, tile_dim2=0
  g1[5] = (int)(unsigned)(stride_elems & 0xFFFFFFFFull);   // dim0_stride[31:0]
  g1[6] = (int)(unsigned)((stride_elems >> 32) & 0xFFFFull);
  g1[7] = 0;
  v4i z4 = {0, 0, 0, 0};
  v8i z8 = {0, 0, 0, 0, 0, 0, 0, 0};
  __builtin_amdgcn_tensor_load_to_lds(g0, g1, z4, z4, z8, 0);
}

// ---------------------------------------------------------------- conversions
__global__ void cvt_bf16_kernel(const float* __restrict__ src,
                                __bf16* __restrict__ dst, int n) {
  for (int i = blockIdx.x * blockDim.x + threadIdx.x; i < n;
       i += gridDim.x * blockDim.x)
    dst[i] = (__bf16)src[i];
}

// W[N,K] fp32 -> WT[K,N] bf16 (coalesced writes; one-time, L2-resident)
__global__ void cvt_w_transpose_kernel(const float* __restrict__ src,
                                       __bf16* __restrict__ dst) {
  const int idx = blockIdx.x * blockDim.x + threadIdx.x;
  const int n = idx & (DM - 1);
  const int k = idx >> 11;
  dst[(size_t)k * DM + n] = (__bf16)src[(size_t)n * DM + k];
}

// ---------------------------------------------------------------- RoPE
// Q layout [B*H*S, 128]: thread handles pair (i, i+64), i in [0,64)
__global__ void rope_q_kernel(__bf16* __restrict__ buf, int total) {
  int idx = blockIdx.x * blockDim.x + threadIdx.x;
  if (idx >= total) return;
  const int i = idx & 63;
  const int s = (idx >> 6) & (SEQ - 1);
  const size_t row = (size_t)(idx >> 6) * HDIM;
  const float ang = (float)s * __expf(-(float)i * 0.14391156831212787f);
  float sn, cs;
  __sincosf(ang, &sn, &cs);
  const float x1 = (float)buf[row + i];
  const float x2 = (float)buf[row + 64 + i];
  buf[row + i]      = (__bf16)(x1 * cs - x2 * sn);
  buf[row + 64 + i] = (__bf16)(x2 * cs + x1 * sn);
}

// K layout transposed [B*H, 128, SEQ]
__global__ void rope_kt_kernel(__bf16* __restrict__ buf, int total) {
  int idx = blockIdx.x * blockDim.x + threadIdx.x;
  if (idx >= total) return;
  const int s = idx & (SEQ - 1);
  const int i = (idx >> 11) & 63;
  const int bh = idx >> 17;
  const size_t base = (size_t)bh * HDIM * SEQ;
  const float ang = (float)s * __expf(-(float)i * 0.14391156831212787f);
  float sn, cs;
  __sincosf(ang, &sn, &cs);
  const float x1 = (float)buf[base + (size_t)i * SEQ + s];
  const float x2 = (float)buf[base + (size_t)(i + 64) * SEQ + s];
  buf[base + (size_t)i * SEQ + s]        = (__bf16)(x1 * cs - x2 * sn);
  buf[base + (size_t)(i + 64) * SEQ + s] = (__bf16)(x2 * cs + x1 * sn);
}

// ---------------------------------------------------------------- GEMM core
// One wave computes a 64x16 fp32 tile of X[M,2048] @ WT[2048,N] (WT = W^T).
// A frag: lane&15 = row, K = (lane>>4)*8 + e(e<8) / +16+(e-8) -> two b128 loads.
// B frag: lane&15 = N col, K order same as A frag -> strided scalar loads.
__device__ __forceinline__ void gemm_tile_64x16(const __bf16* __restrict__ X,
                                                const __bf16* __restrict__ WT,
                                                int m0, int n0, F32x8 acc[4]) {
  const int lane = threadIdx.x & 31;
  const int lr = lane & 15, hi = lane >> 4;
#pragma unroll
  for (int g = 0; g < 4; ++g)
#pragma unroll
    for (int r = 0; r < 8; ++r) acc[g].f[r] = 0.f;
  const __bf16* xbase = X + (size_t)(m0 + lr) * DM + hi * 8;
  const __bf16* wbase = WT + (size_t)(hi * 8) * DM + n0 + lr;
  for (int k0 = 0; k0 < DM; k0 += 32) {
    BF16x16 b;
    const __bf16* ws = wbase + (size_t)k0 * DM;
#pragma unroll
    for (int e = 0; e < 8; ++e) {
      b.h[e] = ws[(size_t)e * DM];
      b.h[e + 8] = ws[(size_t)(e + 16) * DM];
    }
#pragma unroll
    for (int g = 0; g < 4; ++g) {
      BF16x16 a;
      const uint4* xs = (const uint4*)(xbase + (size_t)g * 16 * DM + k0);
      a.u4[0] = xs[0];   // K = k0 + hi*8 .. +7
      a.u4[1] = xs[2];   // K = k0 + hi*8 + 16 .. +23
      acc[g].v = wmma_bf16(a.v, b.v, acc[g].v);
    }
  }
}

// ---------------------------------------------------------------- fused QKV
// grid: (4096/64=64, 3*2048/16=384), block 32. K goes out transposed.
__global__ void __launch_bounds__(32)
qkv_gemm_kernel(const __bf16* __restrict__ X,
                const __bf16* __restrict__ WqT, const __bf16* __restrict__ WkT,
                const __bf16* __restrict__ WvT,
                __bf16* __restrict__ Q, __bf16* __restrict__ KT,
                __bf16* __restrict__ V) {
  const int m0 = blockIdx.x * 64;
  const int nall = blockIdx.y * 16;
  const int which = nall >> 11;
  const int n0 = nall & (DM - 1);
  const __bf16* WT = (which == 0) ? WqT : (which == 1) ? WkT : WvT;
  F32x8 acc[4];
  gemm_tile_64x16(X, WT, m0, n0, acc);
  const int lane = threadIdx.x & 31, lr = lane & 15, hi = lane >> 4;
  const int n = n0 + lr, h = n >> 7, d = n & (HDIM - 1);
#pragma unroll
  for (int g = 0; g < 4; ++g)
#pragma unroll
    for (int r = 0; r < 8; ++r) {   // C frag: row = r + 8*hi, col = lr
      const int m = m0 + g * 16 + r + hi * 8;
      const int bb = m >> 11, s = m & (SEQ - 1);
      const __bf16 val = (__bf16)acc[g].f[r];
      if (which == 1)   // K transposed: [B,H,D,S]
        KT[(((size_t)bb * HEADS + h) * HDIM + d) * SEQ + s] = val;
      else if (which == 0)
        Q[(((size_t)bb * HEADS + h) * SEQ + s) * HDIM + d] = val;
      else
        V[(((size_t)bb * HEADS + h) * SEQ + s) * HDIM + d] = val;
    }
}

// ---------------------------------------------------------------- output proj
__global__ void __launch_bounds__(32)
oproj_gemm_kernel(const __bf16* __restrict__ X, const __bf16* __restrict__ WoT,
                  float* __restrict__ Out) {
  const int m0 = blockIdx.x * 64;
  const int n0 = blockIdx.y * 16;
  F32x8 acc[4];
  gemm_tile_64x16(X, WoT, m0, n0, acc);
  const int lane = threadIdx.x & 31, lr = lane & 15, hi = lane >> 4;
#pragma unroll
  for (int g = 0; g < 4; ++g)
#pragma unroll
    for (int r = 0; r < 8; ++r)
      Out[(size_t)(m0 + g * 16 + r + hi * 8) * DM + n0 + lr] = acc[g].f[r];
}

// ---------------------------------------------------------------- flash attn
// grid: (S/16=128, B*H=32), block 32 (one wave). K^T tiles arrive via TDM,
// double-buffered on TENSORcnt; V fragments load straight from global.
__global__ void __launch_bounds__(32)
flash_attn_kernel(const __bf16* __restrict__ Q, const __bf16* __restrict__ KT,
                  const __bf16* __restrict__ V, __bf16* __restrict__ Attn) {
  __shared__ __bf16 ktile[2][HDIM * 32];  // 2 x 8KB: [d 0..127][key 0..31]
  __shared__ __bf16 plds[16 * 32];        // P relayout (C frag -> A frag)
  const int bh = blockIdx.y;
  const int q0 = blockIdx.x * 16;
  const int b = bh >> 4, h = bh & 15;
  const size_t baseR = (size_t)bh * SEQ * HDIM;   // row-major Q/V base
  const __bf16* ktb = KT + (size_t)bh * HDIM * SEQ;
  const int lane = threadIdx.x & 31, lr = lane & 15, hi = lane >> 4;

  // Q A-fragments (4 chunks of K=32 over head_dim)
  BF16x16 aq[4];
  {
    const __bf16* qrow = Q + baseR + (size_t)(q0 + lr) * HDIM + hi * 8;
#pragma unroll
    for (int dc = 0; dc < 4; ++dc) {
      const uint4* p = (const uint4*)(qrow + dc * 32);
      aq[dc].u4[0] = p[0];
      aq[dc].u4[1] = p[2];
    }
  }

  F32x8 acc[8];
  float m8[8], l8[8];
#pragma unroll
  for (int f = 0; f < 8; ++f)
#pragma unroll
    for (int r = 0; r < 8; ++r) acc[f].f[r] = 0.f;
#pragma unroll
  for (int r = 0; r < 8; ++r) { m8[r] = -1e30f; l8[r] = 0.f; }

  const float scale = 0.088388347648318447f;  // 1/sqrt(128)
  int lo = q0 - (WINDOW - 1); if (lo < 0) lo = 0;
  const int kb_start = lo & ~31;
  const int kb_end = q0 + 15;

  // Prime the TDM pipeline: tile = KT[0..127][kb..kb+31], stride SEQ.
  tdm_load_2d(lds_off_of(&ktile[0][0]), ktb + kb_start, 32, HDIM, SEQ);
  int db = 0;

  for (int kb = kb_start; kb <= kb_end; kb += 32) {
    const int kn = kb + 32;
    if (kn <= kb_end) {
      tdm_load_2d(lds_off_of(&ktile[db ^ 1][0]), ktb + kn, 32, HDIM, SEQ);
      __builtin_amdgcn_s_wait_tensorcnt(1);  // current buffer complete
    } else {
      __builtin_amdgcn_s_wait_tensorcnt(0);
    }
    compiler_fence();
    const __bf16* kt_lds = &ktile[db][0];

    // V B-fragments: lane&15 = dim col f*16+lr, elems = keys in A-frag K order
    BF16x16 bv[8];
    {
      const __bf16* vs = V + baseR + (size_t)(kb + hi * 8) * HDIM + lr;
#pragma unroll
      for (int f = 0; f < 8; ++f)
#pragma unroll
        for (int e = 0; e < 8; ++e) {
          bv[f].h[e] = vs[(size_t)e * HDIM + f * 16];
          bv[f].h[e + 8] = vs[(size_t)(e + 16) * HDIM + f * 16];
        }
    }

    // Scores S[16q x 32k] = Q . K^T : B frag lane&15 = key col, elems = d in A-frag K order
    F32x8 c0, c1;
#pragma unroll
    for (int r = 0; r < 8; ++r) { c0.f[r] = 0.f; c1.f[r] = 0.f; }
#pragma unroll
    for (int dc = 0; dc < 4; ++dc) {
      BF16x16 bk0, bk1;
      const __bf16* kp = kt_lds + (size_t)(dc * 32 + hi * 8) * 32 + lr;
#pragma unroll
      for (int e = 0; e < 8; ++e) {
        bk0.h[e] = kp[e * 32];
        bk0.h[e + 8] = kp[(e + 16) * 32];
        bk1.h[e] = kp[e * 32 + 16];
        bk1.h[e + 8] = kp[(e + 16) * 32 + 16];
      }
      c0.v = wmma_bf16(aq[dc].v, bk0.v, c0.v);
      c1.v = wmma_bf16(aq[dc].v, bk1.v, c1.v);
    }

    // Mask + online softmax (C-frag row r lives in one 16-lane half)
    const int key0 = kb + lr, key1 = kb + 16 + lr;
    const int qrow0 = q0 + hi * 8;
    float p0[8], p1[8], alph[8];
#pragma unroll
    for (int r = 0; r < 8; ++r) {
      const int qi = qrow0 + r;
      float s0 = c0.f[r] * scale;
      float s1 = c1.f[r] * scale;
      if (key0 > qi || (qi - key0) >= WINDOW) s0 = -1e30f;
      if (key1 > qi || (qi - key1) >= WINDOW) s1 = -1e30f;
      float mx = fmaxf(s0, s1);
#pragma unroll
      for (int off = 8; off >= 1; off >>= 1)
        mx = fmaxf(mx, __shfl_xor(mx, off, 16));
      const float mnew = fmaxf(m8[r], mx);
      alph[r] = __expf(m8[r] - mnew);
      p0[r] = (s0 <= -1e29f) ? 0.f : __expf(s0 - mnew);
      p1[r] = (s1 <= -1e29f) ? 0.f : __expf(s1 - mnew);
      float rs = p0[r] + p1[r];
#pragma unroll
      for (int off = 8; off >= 1; off >>= 1)
        rs += __shfl_xor(rs, off, 16);
      l8[r] = l8[r] * alph[r] + rs;
      m8[r] = mnew;
    }
#pragma unroll
    for (int f = 0; f < 8; ++f)
#pragma unroll
      for (int r = 0; r < 8; ++r) acc[f].f[r] *= alph[r];

    // Relayout P (C frag -> A frag) through LDS
#pragma unroll
    for (int r = 0; r < 8; ++r) {
      plds[(r + hi * 8) * 32 + lr]      = (__bf16)p0[r];
      plds[(r + hi * 8) * 32 + 16 + lr] = (__bf16)p1[r];
    }
    lds_sync();
    BF16x16 ap;
#pragma unroll
    for (int e = 0; e < 8; ++e) {
      ap.h[e]     = plds[lr * 32 + hi * 8 + e];
      ap.h[e + 8] = plds[lr * 32 + hi * 8 + 16 + e];
    }
    lds_sync();

    // O += P . V
#pragma unroll
    for (int f = 0; f < 8; ++f)
      acc[f].v = wmma_bf16(ap.v, bv[f].v, acc[f].v);

    db ^= 1;
  }

  // Normalize, scatter to attn buffer [B*S, 2048]
#pragma unroll
  for (int f = 0; f < 8; ++f)
#pragma unroll
    for (int r = 0; r < 8; ++r) {
      const int s = q0 + r + hi * 8;
      Attn[((size_t)b * SEQ + s) * DM + h * HDIM + f * 16 + lr] =
          (__bf16)(acc[f].f[r] / l8[r]);
    }
}

// ---------------------------------------------------------------- launch
extern "C" void kernel_launch(void* const* d_in, const int* in_sizes, int n_in,
                              void* d_out, int out_size, void* d_ws, size_t ws_size,
                              hipStream_t stream) {
  (void)in_sizes; (void)n_in; (void)out_size; (void)ws_size;
  const float* hs = (const float*)d_in[0];
  const float* Wq = (const float*)d_in[1];
  const float* Wk = (const float*)d_in[2];
  const float* Wv = (const float*)d_in[3];
  const float* Wo = (const float*)d_in[4];
  float* out = (float*)d_out;

  const size_t nHid = (size_t)2 * SEQ * DM;             // 8,388,608
  const size_t nW   = (size_t)DM * DM;                  // 4,194,304
  const size_t nQKV = (size_t)2 * HEADS * SEQ * HDIM;   // 8,388,608

  __bf16* p    = (__bf16*)d_ws;   // ~117 MB bf16 staging
  __bf16* hb   = p; p += nHid;
  __bf16* wqT  = p; p += nW;
  __bf16* wkT  = p; p += nW;
  __bf16* wvT  = p; p += nW;
  __bf16* woT  = p; p += nW;
  __bf16* qb   = p; p += nQKV;
  __bf16* ktb  = p; p += nQKV;
  __bf16* vb   = p; p += nQKV;
  __bf16* ab   = p; p += nHid;

  cvt_bf16_kernel<<<4096, 256, 0, stream>>>(hs, hb, (int)nHid);
  cvt_w_transpose_kernel<<<(int)(nW / 256), 256, 0, stream>>>(Wq, wqT);
  cvt_w_transpose_kernel<<<(int)(nW / 256), 256, 0, stream>>>(Wk, wkT);
  cvt_w_transpose_kernel<<<(int)(nW / 256), 256, 0, stream>>>(Wv, wvT);
  cvt_w_transpose_kernel<<<(int)(nW / 256), 256, 0, stream>>>(Wo, woT);

  qkv_gemm_kernel<<<dim3(64, 384), dim3(32), 0, stream>>>(hb, wqT, wkT, wvT,
                                                          qb, ktb, vb);

  const int ropeTotal = 2 * HEADS * SEQ * 64;  // 4,194,304
  rope_q_kernel<<<(ropeTotal + 255) / 256, 256, 0, stream>>>(qb, ropeTotal);
  rope_kt_kernel<<<(ropeTotal + 255) / 256, 256, 0, stream>>>(ktb, ropeTotal);

  flash_attn_kernel<<<dim3(128, 32), dim3(32), 0, stream>>>(qb, ktb, vb, ab);

  oproj_gemm_kernel<<<dim3(64, 128), dim3(32), 0, stream>>>(ab, woT, out);
}
